// H3GNN_64244120814024
// MI455X (gfx1250) — matrix operand order repaired and checked by reading, not compiled, not run on hardware
//
#include <hip/hip_runtime.h>
#include <hip/hip_bf16.h>
#include <math.h>

// ---------------- CDNA5 WMMA types ----------------
typedef _Float16 half_t;
typedef __attribute__((ext_vector_type(16))) _Float16 v16h;
typedef __attribute__((ext_vector_type(8)))  _Float16 v8h;
typedef __attribute__((ext_vector_type(8)))  float    v8f;

#define Bdim 128
#define Sdim 512
#define Hdim 256
#define H2   512   // 2H
#define H3   768   // 3H

__device__ __forceinline__ v8f wmma_f16(v16h a, v16h b, v8f c) {
    // D = A(16x32 f16) * B(32x16 f16) + C(16x16 f32)
    return __builtin_amdgcn_wmma_f32_16x16x32_f16(
        false, a, false, b, (short)0, c, false, false);
}

// A-operand fragment from f32 row-major [m][k] matrix (ld in floats).
// Lane layout (16-bit A, 16x32): lanes 0-15: K {0..7, 16..23}; lanes 16-31: K {8..15, 24..31}.
__device__ __forceinline__ v16h frag_a_f32(const float* A, int ld, int m, int k0, int lane) {
    int hs = lane >> 4;
    const float* p1 = A + (size_t)m * ld + k0 + hs * 8;
    const float* p2 = p1 + 16;
    v8f f1 = *(const v8f*)p1;
    v8f f2 = *(const v8f*)p2;
    v16h a;
#pragma unroll
    for (int i = 0; i < 8; i++) { a[i] = (_Float16)f1[i]; a[8 + i] = (_Float16)f2[i]; }
    return a;
}

// A-operand fragment from f16 row-major [m][k] matrix (ld in halves).
__device__ __forceinline__ v16h frag_a_f16(const half_t* A, int ld, int m, int k0, int lane) {
    int hs = lane >> 4;
    const half_t* p1 = A + (size_t)m * ld + k0 + hs * 8;
    const half_t* p2 = p1 + 16;
    v8h l = *(const v8h*)p1;
    v8h h = *(const v8h*)p2;
    v16h a;
#pragma unroll
    for (int i = 0; i < 8; i++) { a[i] = l[i]; a[8 + i] = h[i]; }
    return a;
}

// B-operand fragment from f16 K-contiguous [n][k] matrix (ld in halves).
// Lane layout (16-bit B, 32x16): lanes 0-15: K {0..7, 16..23}; lanes 16-31: K {8..15, 24..31}.
__device__ __forceinline__ v16h frag_b_f16(const half_t* Bm, int ld, int n, int k0, int lane) {
    const half_t* p = Bm + (size_t)n * ld + k0 + (lane >> 4) * 8;
    v8h l = *(const v8h*)p;
    v8h h = *(const v8h*)(p + 16);
    v16h b;
#pragma unroll
    for (int i = 0; i < 8; i++) { b[i] = l[i]; b[8 + i] = h[i]; }
    return b;
}

// ============ Kernel 0: one-time f32 -> f16 conversion of all weight matrices ============
__global__ __launch_bounds__(256)
void k0_cvt(const float* __restrict__ W_in, const float* __restrict__ W_out,
            const float* __restrict__ w_ih, const float* __restrict__ w_hh,
            half_t* __restrict__ W_in16, half_t* __restrict__ W_out16,
            half_t* __restrict__ w_ih16, half_t* __restrict__ w_hh16) {
    const int nWin = Hdim * Hdim;          // 65536
    const int nIh  = H3 * H2;              // 393216
    const int nHh  = H3 * Hdim;            // 196608
    const int total = nWin * 2 + nIh + nHh;
    for (int idx = blockIdx.x * blockDim.x + threadIdx.x; idx < total;
         idx += gridDim.x * blockDim.x) {
        int i = idx;
        if (i < nWin)            { W_in16[i]  = (half_t)W_in[i];  continue; }
        i -= nWin;
        if (i < nWin)            { W_out16[i] = (half_t)W_out[i]; continue; }
        i -= nWin;
        if (i < nIh)             { w_ih16[i]  = (half_t)w_ih[i];  continue; }
        i -= nIh;
        w_hh16[i] = (half_t)w_hh[i];
    }
}

// ============ Kernel 1: h_in/h_out = hidden @ W^T + b, stored transposed f16 ============
// Output h_*T[b][g][s]  (g-major so K2's B-operand is K=j contiguous)
__global__ __launch_bounds__(256)
void k1_edge(const float* __restrict__ hidden,
             const half_t* __restrict__ W_in16,  const float* __restrict__ b_in,
             const half_t* __restrict__ W_out16, const float* __restrict__ b_out,
             half_t* __restrict__ h_inT, half_t* __restrict__ h_outT) {
    int rt   = blockIdx.x;                 // 16-row tile of the 65536 rows
    int wave = threadIdx.x >> 5;
    int lane = threadIdx.x & 31;
    int m    = lane & 15;
    int hs   = lane >> 4;
    int row0  = rt << 4;
    int batch = row0 >> 9;                 // / S
    int s0    = row0 & (Sdim - 1);

    const float* Arow = hidden + (size_t)row0 * Hdim;
    // hoist all A fragments (K = 256 -> 8 frags, shared by 4 output tiles)
    v16h af[8];
#pragma unroll
    for (int kk = 0; kk < 8; kk++) af[kk] = frag_a_f32(Arow, Hdim, m, kk * 32, lane);

#pragma unroll
    for (int i = 0; i < 4; i++) {
        int t   = wave * 4 + i;            // 32 tiles: 2 weights x 16 col-tiles
        int mat = t >> 4;
        int n0  = (t & 15) << 4;
        const half_t* W    = mat ? W_out16 : W_in16;
        const float*  bias = mat ? b_out   : b_in;
        half_t*       dstT = mat ? h_outT  : h_inT;
        // stage all B fragments (distinct registers -> partial loadcnt waits)
        v16h bf[8];
#pragma unroll
        for (int kk = 0; kk < 8; kk++) bf[kk] = frag_b_f16(W, Hdim, n0 + m, kk * 32, lane);
        v8f c = {};
#pragma unroll
        for (int kk = 0; kk < 8; kk++) c = wmma_f16(af[kk], bf[kk], c);

        int g = n0 + m;
        float bv = bias[g];
        v8h o;
#pragma unroll
        for (int j = 0; j < 8; j++) o[j] = (half_t)(c[j] + bv);
        // transposed store: 8 consecutive s at fixed g -> one b128 store
        half_t* dst = dstT + ((size_t)batch * Hdim + g) * Sdim + s0 + hs * 8;
        *(v8h*)dst = o;
    }
}

// ============ Kernel 2: input_in/out = A[:, :, :S|S:] @ h_* + bias, f16 row-major ============
// inputs[(b*S + i)][ mat*H + g ]
__global__ __launch_bounds__(256)
void k2_adj(const float* __restrict__ Amat,
            const half_t* __restrict__ h_inT, const half_t* __restrict__ h_outT,
            const float* __restrict__ b_iah,  const float* __restrict__ b_oah,
            half_t* __restrict__ inputs) {
    int bid   = blockIdx.x;
    int batch = bid >> 5;                  // / (S/16)
    int it    = bid & 31;
    int i0    = it << 4;
    int wave = threadIdx.x >> 5;
    int lane = threadIdx.x & 31;
    int m    = lane & 15;
    int hs   = lane >> 4;

    int mat   = wave & 1;                  // 0: in-part, 1: out-part
    int nbase = (wave >> 1) << 4;          // {0,16,32,48}, +q*64 covers all 16 tiles

    const float*  Arow = Amat + ((size_t)batch * Sdim + i0) * (2 * Sdim) + mat * Sdim;
    const half_t* hT   = (mat ? h_outT : h_inT) + (size_t)batch * Hdim * Sdim;
    const float*  bias = mat ? b_oah : b_iah;

    v8f c[4] = {{}, {}, {}, {}};
    for (int ks = 0; ks < 16; ks++) {
        int k0 = ks * 32;
        // pull the next K-chunk of the streamed A matrix toward the WGP
        if (ks < 15)
            __builtin_prefetch(Arow + (size_t)m * (2 * Sdim) + k0 + 32, 0, 0);
        // stage A + all 4 B fragments before any WMMA: 12 b128 loads in flight,
        // so waits before the WMMAs are partial instead of loadcnt==0 drains
        v16h a = frag_a_f32(Arow, 2 * Sdim, m, k0, lane);
        v16h bf[4];
#pragma unroll
        for (int q = 0; q < 4; q++)
            bf[q] = frag_b_f16(hT, Sdim, nbase + q * 64 + m, k0, lane);
#pragma unroll
        for (int q = 0; q < 4; q++)
            c[q] = wmma_f16(a, bf[q], c[q]);
    }
#pragma unroll
    for (int q = 0; q < 4; q++) {
        int g = nbase + q * 64 + m;
        float bv = bias[g];
        size_t col = (size_t)mat * Hdim + g;
#pragma unroll
        for (int j = 0; j < 8; j++) {
            int s = i0 + hs * 8 + j;
            inputs[((size_t)batch * Sdim + s) * H2 + col] = (half_t)(c[q][j] + bv);
        }
    }
}

// ============ Kernel 3: gi + gh + fused GRU gates ============
__global__ __launch_bounds__(256)
void k3_gru(const half_t* __restrict__ inputs, const float* __restrict__ hidden,
            const half_t* __restrict__ w_ih16, const float* __restrict__ b_ih,
            const half_t* __restrict__ w_hh16, const float* __restrict__ b_hh,
            float* __restrict__ out) {
    __shared__ float sum_ri[16][H2];   // (i_r+h_r | i_i+h_i)  32 KB
    __shared__ float lds_in[16][Hdim]; // i_n                  16 KB
    __shared__ float lds_hn[16][Hdim]; // h_n                  16 KB

    int rt   = blockIdx.x;
    int row0 = rt << 4;
    int wave = threadIdx.x >> 5;
    int lane = threadIdx.x & 31;
    int m    = lane & 15;
    int hs   = lane >> 4;

    const half_t* inRow  = inputs + (size_t)row0 * H2;
    const float*  hidRow = hidden + (size_t)row0 * Hdim;

    // wave owns 6 column tiles t = wave + 8*i of the 48 (3H/16);
    // K-outer loop: each A fragment loaded once, reused 6x; all 6 B fragments
    // staged into distinct registers before the WMMA burst.
    v8f cg[6] = {{}, {}, {}, {}, {}, {}};
    v8f ch[6] = {{}, {}, {}, {}, {}, {}};

    for (int ks = 0; ks < 16; ks++) {          // gi: K = 2H = 512
        int k0 = ks * 32;
        v16h a = frag_a_f16(inRow, H2, m, k0, lane);
        v16h bf[6];
#pragma unroll
        for (int i = 0; i < 6; i++)
            bf[i] = frag_b_f16(w_ih16, H2, (wave + 8 * i) * 16 + m, k0, lane);
#pragma unroll
        for (int i = 0; i < 6; i++)
            cg[i] = wmma_f16(a, bf[i], cg[i]);
    }
    for (int ks = 0; ks < 8; ks++) {           // gh: K = H = 256
        int k0 = ks * 32;
        v16h a = frag_a_f32(hidRow, Hdim, m, k0, lane);
        v16h bf[6];
#pragma unroll
        for (int i = 0; i < 6; i++)
            bf[i] = frag_b_f16(w_hh16, Hdim, (wave + 8 * i) * 16 + m, k0, lane);
#pragma unroll
        for (int i = 0; i < 6; i++)
            ch[i] = wmma_f16(a, bf[i], ch[i]);
    }

#pragma unroll
    for (int i = 0; i < 6; i++) {
        int t  = wave + 8 * i;
        int g3 = t * 16 + m;
        float bi = b_ih[g3];
        float bh = b_hh[g3];
        if (t < 32) {                      // r / i thirds: only the sum is needed
#pragma unroll
            for (int j = 0; j < 8; j++)
                sum_ri[hs * 8 + j][g3] = cg[i][j] + ch[i][j] + bi + bh;
        } else {                           // n third: keep i_n, h_n separate
            int h = g3 - H2;
#pragma unroll
            for (int j = 0; j < 8; j++) {
                lds_in[hs * 8 + j][h] = cg[i][j] + bi;
                lds_hn[hs * 8 + j][h] = ch[i][j] + bh;
            }
        }
    }
    __syncthreads();

    // gate combine: 16 rows x 256 cols, one col per thread per row
    int h = threadIdx.x;                   // 0..255
    for (int s = 0; s < 16; s++) {
        float rg = 1.0f / (1.0f + expf(-sum_ri[s][h]));
        float ig = 1.0f / (1.0f + expf(-sum_ri[s][Hdim + h]));
        float ng = tanhf(lds_in[s][h] + rg * lds_hn[s][h]);
        size_t idx = (size_t)(row0 + s) * Hdim + h;
        float hv = hidden[idx];
        out[idx] = hv + ig * (ng - hv);
    }
}

// ---------------- launch ----------------
extern "C" void kernel_launch(void* const* d_in, const int* in_sizes, int n_in,
                              void* d_out, int out_size, void* d_ws, size_t ws_size,
                              hipStream_t stream) {
    const float* Amat   = (const float*)d_in[0];
    const float* hidden = (const float*)d_in[1];
    // d_in[2] = mask (unused by reference with issrgnn path)
    const float* W_in  = (const float*)d_in[3];
    const float* b_in  = (const float*)d_in[4];
    const float* W_out = (const float*)d_in[5];
    const float* b_out = (const float*)d_in[6];
    const float* b_iah = (const float*)d_in[7];
    const float* b_oah = (const float*)d_in[8];
    const float* w_ih  = (const float*)d_in[9];
    const float* b_ih  = (const float*)d_in[10];
    const float* w_hh  = (const float*)d_in[11];
    const float* b_hh  = (const float*)d_in[12];
    float* out = (float*)d_out;

    // workspace: h_inT (32MB) | h_outT (32MB) | inputs (64MB) | f16 weights (~1.4MB)
    half_t* h_inT  = (half_t*)d_ws;
    half_t* h_outT = h_inT  + (size_t)Bdim * Hdim * Sdim;
    half_t* inputs = h_outT + (size_t)Bdim * Hdim * Sdim;
    half_t* W_in16  = inputs + (size_t)Bdim * Sdim * H2;
    half_t* W_out16 = W_in16  + (size_t)Hdim * Hdim;
    half_t* w_ih16  = W_out16 + (size_t)Hdim * Hdim;
    half_t* w_hh16  = w_ih16  + (size_t)H3 * H2;

    const int rowTiles = (Bdim * Sdim) / 16;   // 4096

    k0_cvt <<<512, 256, 0, stream>>>(W_in, W_out, w_ih, w_hh,
                                     W_in16, W_out16, w_ih16, w_hh16);
    k1_edge<<<rowTiles, 256, 0, stream>>>(hidden, W_in16, b_in, W_out16, b_out,
                                          h_inT, h_outT);
    k2_adj <<<rowTiles, 256, 0, stream>>>(Amat, h_inT, h_outT, b_iah, b_oah, inputs);
    k3_gru <<<rowTiles, 256, 0, stream>>>(inputs, hidden, w_ih16, b_ih, w_hh16, b_hh, out);
}
